// MatchNet_73967926772216
// MI455X (gfx1250) — matrix operand order repaired and checked by reading, not compiled, not run on hardware
//
#include <hip/hip_runtime.h>
#include <hip/hip_bf16.h>

#define EPS_F      1e-5f
#define TEMP_F     3.0f
#define MASKFILL_F 1e-4f

typedef __attribute__((ext_vector_type(16))) _Float16 v16h;
typedef __attribute__((ext_vector_type(8)))  _Float16 v8h;
typedef __attribute__((ext_vector_type(8)))  float    v8f;

// ---------------- mutual matching: row max (over s) ----------------
// grid = B*1024 rows, 256 threads; each row is 1024 contiguous floats
__global__ __launch_bounds__(256) void k_rowmax(const float* __restrict__ x,
                                                float* __restrict__ mA) {
  __shared__ float red[256];
  int row = blockIdx.x;
  const float* p = x + ((size_t)row << 10);
  float m = -3.4e38f;
  for (int j = threadIdx.x; j < 1024; j += 256) m = fmaxf(m, p[j]);
  red[threadIdx.x] = m;
  __syncthreads();
  for (int s = 128; s > 0; s >>= 1) {
    if (threadIdx.x < s) red[threadIdx.x] = fmaxf(red[threadIdx.x], red[threadIdx.x + s]);
    __syncthreads();
  }
  if (threadIdx.x == 0) mA[row] = red[0];
}

// ---------------- mutual matching: col max (over q) ----------------
// one thread per (b,s); strided loads are coalesced across threads
__global__ __launch_bounds__(256) void k_colmax(const float* __restrict__ x,
                                                float* __restrict__ mB) {
  int idx = blockIdx.x * 256 + threadIdx.x;   // b*1024 + s
  int b = idx >> 10, s = idx & 1023;
  const float* p = x + ((size_t)b << 20) + s;
  float m = -3.4e38f;
  for (int i = 0; i < 1024; ++i) m = fmaxf(m, p[(size_t)i << 10]);
  mB[idx] = m;
}

// ---------------- mutual matching: pointwise ----------------
__global__ __launch_bounds__(256) void k_mutual(const float* __restrict__ x,
                                                const float* __restrict__ mA,
                                                const float* __restrict__ mB,
                                                float* __restrict__ y) {
  size_t idx = (size_t)blockIdx.x * 256 + threadIdx.x;
  int b = (int)(idx >> 20);
  int i = (int)((idx >> 10) & 1023);
  int s = (int)(idx & 1023);
  float v = x[idx];
  y[idx] = v * (v / (mA[(b << 10) + i] + EPS_F)) * (v / (mB[(b << 10) + s] + EPS_F));
}

// ---------------- CP conv4d + relu (+optional accumulate) ----------------
// block = one (b,co,ha,wa); 256 threads cover the 1024 (hb,wb) outputs, 4 each.
// conv1 (sw1) runs over (ha,wa); conv2 (sw2) over (hb,wb). Swap weight args for
// the transposed stack.
__global__ __launch_bounds__(256) void k_cpconv(const float* __restrict__ x,
                                                const float* __restrict__ w1,
                                                const float* __restrict__ b1,
                                                const float* __restrict__ w2,
                                                const float* __restrict__ b2,
                                                float* __restrict__ y,
                                                int Ci, int Co, int accumulate) {
  __shared__ float sw1[96];
  __shared__ float sw2[96];
  int z = blockIdx.x;
  int wa = z & 31; z >>= 5;
  int ha = z & 31; z >>= 5;
  int co = z % Co;
  int b  = z / Co;
  int tid = threadIdx.x;
  int nw = Ci * 9;
  for (int t = tid; t < nw; t += 256) {
    sw1[t] = w1[co * nw + t];
    sw2[t] = w2[co * nw + t];
  }
  __syncthreads();
  float bias = b1[co] + b2[co];
  const float* xb = x + (((size_t)b * Ci) << 20);
  size_t plane = ((size_t)(ha * 32 + wa)) << 10;
  size_t outBase = (((size_t)(b * Co + co)) << 20) + plane;
  for (int jj = 0; jj < 4; ++jj) {
    int j = tid + (jj << 8);
    int hb = j >> 5, wb = j & 31;
    float acc = bias;
    for (int ci = 0; ci < Ci; ++ci) {
      const float* xc = xb + ((size_t)ci << 20);
      const float* s1 = sw1 + ci * 9;
      const float* s2 = sw2 + ci * 9;
#pragma unroll
      for (int dh = -1; dh <= 1; ++dh) {
        int ha2 = ha + dh;
        if ((unsigned)ha2 >= 32u) continue;
#pragma unroll
        for (int dw = -1; dw <= 1; ++dw) {
          int wa2 = wa + dw;
          if ((unsigned)wa2 >= 32u) continue;
          acc += s1[(dh + 1) * 3 + (dw + 1)] * xc[(((size_t)(ha2 * 32 + wa2)) << 10) + j];
        }
      }
      const float* xp = xc + plane;
#pragma unroll
      for (int dh = -1; dh <= 1; ++dh) {
        int hb2 = hb + dh;
        if ((unsigned)hb2 >= 32u) continue;
#pragma unroll
        for (int dw = -1; dw <= 1; ++dw) {
          int wb2 = wb + dw;
          if ((unsigned)wb2 >= 32u) continue;
          acc += s2[(dh + 1) * 3 + (dw + 1)] * xp[(hb2 << 5) + wb2];
        }
      }
    }
    float r = fmaxf(acc, 0.0f);
    if (accumulate) y[outBase + j] += r;
    else            y[outBase + j] = r;
  }
}

// ---------------- masked softmax row -> f16 probabilities ----------------
__global__ __launch_bounds__(256) void k_softmax(const float* __restrict__ c,
                                                 const int* __restrict__ mask,
                                                 _Float16* __restrict__ attn) {
  __shared__ float red[256];
  int row = blockIdx.x;              // b*1024 + q
  int b = row >> 10;
  const float* p = c + ((size_t)row << 10);
  const int* mk = mask + (b << 10);
  float vals[4];
  float m = -3.4e38f;
#pragma unroll
  for (int jj = 0; jj < 4; ++jj) {
    int s = threadIdx.x + (jj << 8);
    float x = mk[s] ? MASKFILL_F : p[s];
    x *= TEMP_F;
    vals[jj] = x;
    m = fmaxf(m, x);
  }
  red[threadIdx.x] = m;
  __syncthreads();
  for (int st = 128; st > 0; st >>= 1) {
    if (threadIdx.x < st) red[threadIdx.x] = fmaxf(red[threadIdx.x], red[threadIdx.x + st]);
    __syncthreads();
  }
  float M = red[0];
  __syncthreads();
  float sum = 0.f;
#pragma unroll
  for (int jj = 0; jj < 4; ++jj) { vals[jj] = __expf(vals[jj] - M); sum += vals[jj]; }
  red[threadIdx.x] = sum;
  __syncthreads();
  for (int st = 128; st > 0; st >>= 1) {
    if (threadIdx.x < st) red[threadIdx.x] += red[threadIdx.x + st];
    __syncthreads();
  }
  float inv = 1.0f / red[0];
#pragma unroll
  for (int jj = 0; jj < 4; ++jj) {
    int s = threadIdx.x + (jj << 8);
    attn[((size_t)row << 10) + s] = (_Float16)(vals[jj] * inv);
  }
}

// ---------------- f32 -> f16 convert ----------------
__global__ __launch_bounds__(256) void k_tohalf(const float* __restrict__ x,
                                                _Float16* __restrict__ y, int n) {
  int i = blockIdx.x * 256 + threadIdx.x;
  if (i < n) y[i] = (_Float16)x[i];
}

// ---------------- batched GEMM: out[b] = V[b](256x1024) * attn[b]^T ----------------
// one wave per 16x16 output tile; K=1024 in 32-steps via v_wmma_f32_16x16x32_f16.
// A-frag lane layout (ISA 16-bit A 16x32): lane r(+16g) holds row M=m0+r,
//   K = {g*8..g*8+7} U {16+g*8..16+g*8+7}  -> two contiguous 16B loads.
// B-frag (32x16): lane r(+16g) holds col N=n0+r, K = {g*8..g*8+7} U {16+g*8..16+g*8+7}
//   -> two contiguous 16B loads from attn row (q=n), since B[k][n] = attn[n][k].
__global__ __launch_bounds__(256) void k_attn_gemm(const _Float16* __restrict__ vh,
                                                   const _Float16* __restrict__ attn,
                                                   float* __restrict__ out) {
  int gtid = blockIdx.x * 256 + threadIdx.x;
  int wave = gtid >> 5;
  int lane = threadIdx.x & 31;
  int b = wave >> 10;           // 1024 tiles per batch (16 x 64)
  int t = wave & 1023;
  int m0 = (t >> 6) << 4;
  int n0 = (t & 63) << 4;
  int r = lane & 15, g = lane >> 4;

  const _Float16* aRow = vh   + ((size_t)b << 18) + ((size_t)(m0 + r) << 10);
  const _Float16* bRow = attn + ((size_t)b << 20) + ((size_t)(n0 + r) << 10);

  v8f acc = {0.f, 0.f, 0.f, 0.f, 0.f, 0.f, 0.f, 0.f};
  for (int k = 0; k < 1024; k += 32) {
    v8h alo = *(const v8h*)(aRow + k + g * 8);
    v8h ahi = *(const v8h*)(aRow + k + 16 + g * 8);
    v8h blo = *(const v8h*)(bRow + k + g * 8);
    v8h bhi = *(const v8h*)(bRow + k + 16 + g * 8);
    v16h a  = __builtin_shufflevector(alo, ahi, 0,1,2,3,4,5,6,7,8,9,10,11,12,13,14,15);
    v16h bb = __builtin_shufflevector(blo, bhi, 0,1,2,3,4,5,6,7,8,9,10,11,12,13,14,15);
    acc = __builtin_amdgcn_wmma_f32_16x16x32_f16(false, a, false, bb,
                                                 (short)0, acc, false, false);
  }
  // C/D layout: VGPR i -> row m0 + i + g*8, col n0 + r (coalesced per half-wave)
  float* C = out + ((size_t)b << 18) + n0 + r;
#pragma unroll
  for (int i = 0; i < 8; ++i) C[(size_t)(m0 + i + g * 8) << 10] = acc[i];
}

extern "C" void kernel_launch(void* const* d_in, const int* in_sizes, int n_in,
                              void* d_out, int out_size, void* d_ws, size_t ws_size,
                              hipStream_t stream) {
  (void)in_sizes; (void)n_in; (void)out_size; (void)ws_size;
  const float* corr = (const float*)d_in[0];   // (4,32,32,32,32)
  const float* v    = (const float*)d_in[1];   // (4,256,1024)
  const int*   mask = (const int*)d_in[2];     // (4,1024)
  const float* w[12];
  for (int i = 0; i < 12; ++i) w[i] = (const float*)d_in[3 + i];
  // layer i: w1=w[4i+0], b1=w[4i+1], w2=w[4i+2], b2=w[4i+3]

  const size_t N4  = (size_t)1 << 20;   // 32^4
  const size_t BN4 = N4 * 4;            // 4,194,304 elems

  float* ws   = (float*)d_ws;
  float* c0   = ws;                     // 16.8 MB
  float* t0   = c0 + BN4;               // 167.8 MB (10-ch ping)
  float* t1   = t0 + 10 * BN4;          // 167.8 MB (10-ch pong)
  float* ysum = t1 + 10 * BN4;          // 16.8 MB
  float* mA   = ysum + BN4;             // 4096 f32
  float* mB   = mA + 4096;              // 4096 f32
  _Float16* attnH = (_Float16*)t0;      // alias t0 (free after conv stacks)
  _Float16* vhalf = attnH + BN4;        // alias inside t0
  float* out = (float*)d_out;

  dim3 B256(256);

  // ---- mutual matching #1: corr -> c0 ----
  k_rowmax<<<4096, B256, 0, stream>>>(corr, mA);
  k_colmax<<<16,   B256, 0, stream>>>(corr, mB);
  k_mutual<<<16384, B256, 0, stream>>>(corr, mA, mB, c0);

  // ---- stack A: normal weight roles ----
  k_cpconv<<<40960, B256, 0, stream>>>(c0, w[0], w[1], w[2], w[3],  t0, 1, 10, 0);
  k_cpconv<<<40960, B256, 0, stream>>>(t0, w[4], w[5], w[6], w[7],  t1, 10, 10, 0);
  k_cpconv<<<4096,  B256, 0, stream>>>(t1, w[8], w[9], w[10], w[11], ysum, 10, 1, 0);

  // ---- stack B: (h,w)<->(hs,ws) transpose == swapped conv roles ----
  k_cpconv<<<40960, B256, 0, stream>>>(c0, w[2], w[3], w[0], w[1],  t0, 1, 10, 0);
  k_cpconv<<<40960, B256, 0, stream>>>(t0, w[6], w[7], w[4], w[5],  t1, 10, 10, 0);
  k_cpconv<<<4096,  B256, 0, stream>>>(t1, w[10], w[11], w[8], w[9], ysum, 10, 1, 1);

  // ---- mutual matching #2: ysum -> c0 ----
  k_rowmax<<<4096, B256, 0, stream>>>(ysum, mA);
  k_colmax<<<16,   B256, 0, stream>>>(ysum, mB);
  k_mutual<<<16384, B256, 0, stream>>>(ysum, mA, mB, c0);

  // ---- masked softmax -> f16 attn; v -> f16 ----
  k_softmax<<<4096, B256, 0, stream>>>(c0, mask, attnH);
  k_tohalf<<<4096, B256, 0, stream>>>(v, vhalf, 4 * 256 * 1024);

  // ---- WMMA batched GEMM -> d_out ----
  k_attn_gemm<<<512, B256, 0, stream>>>(vhalf, attnH, out);
}
